// HSTULayer_38585986187605
// MI455X (gfx1250) — matrix operand order repaired and checked by reading, not compiled, not run on hardware
//
#include <hip/hip_runtime.h>
#include <hip/hip_bf16.h>
#include <math.h>

// ---- problem constants (HSTU layer) ----
#define EMB   512      // E
#define SEQ   2048     // S
#define NBAT  4        // B
#define NH    8        // H
#define AD    64       // A (attention dim per head)
#define LDIM  64       // L (linear dim per head)
#define BS    (NBAT*SEQ)   // 8192 tokens
#define NPROJ 2048         // 2*L*H + 2*A*H

typedef __attribute__((ext_vector_type(16))) __bf16 v16bf;
typedef __attribute__((ext_vector_type(8)))  __bf16 v8bf;
typedef __attribute__((ext_vector_type(8)))  float  v8f;
typedef __attribute__((ext_vector_type(4)))  float  v4f;

__device__ __forceinline__ v8f wmma_bf16(v16bf a, v16bf b, v8f c) {
  return __builtin_amdgcn_wmma_f32_16x16x32_bf16(false, a, false, b, (short)0, c, false, false);
}

__device__ __forceinline__ v16bf cat8(v8bf lo, v8bf hi) {
  return __builtin_shufflevector(lo, hi, 0,1,2,3,4,5,6,7,8,9,10,11,12,13,14,15);
}

// ---------------------------------------------------------------------------
// Kernel 1: input LayerNorm + cast to bf16.  One wave per token row (E=512).
// ---------------------------------------------------------------------------
__global__ __launch_bounds__(256) void ln_cast_kernel(
    const float* __restrict__ x, const float* __restrict__ g,
    const float* __restrict__ be, __bf16* __restrict__ xn) {
  int row  = (blockIdx.x * 256 + threadIdx.x) >> 5;
  int lane = threadIdx.x & 31;
  const float* rp = x + (size_t)row * EMB + lane * 16;
  float v[16];
  float s1 = 0.f, s2 = 0.f;
  #pragma unroll
  for (int i = 0; i < 4; ++i) {
    v4f t = *(const v4f*)(rp + i * 4);
    #pragma unroll
    for (int j = 0; j < 4; ++j) { float f = t[j]; v[i*4+j] = f; s1 += f; s2 += f*f; }
  }
  #pragma unroll
  for (int off = 16; off > 0; off >>= 1) {
    s1 += __shfl_xor(s1, off, 32);
    s2 += __shfl_xor(s2, off, 32);
  }
  float mu   = s1 * (1.f / EMB);
  float rstd = rsqrtf(s2 * (1.f / EMB) - mu * mu + 1e-6f);
  const float* gp = g  + lane * 16;
  const float* bp = be + lane * 16;
  __bf16 ov[16];
  #pragma unroll
  for (int i = 0; i < 4; ++i) {
    v4f gg = *(const v4f*)(gp + i * 4);
    v4f bb = *(const v4f*)(bp + i * 4);
    #pragma unroll
    for (int j = 0; j < 4; ++j)
      ov[i*4+j] = (__bf16)((v[i*4+j] - mu) * rstd * gg[j] + bb[j]);
  }
  v8bf o0, o1;
  #pragma unroll
  for (int j = 0; j < 8; ++j) { o0[j] = ov[j]; o1[j] = ov[8 + j]; }
  v8bf* op = (v8bf*)(xn + (size_t)row * EMB + lane * 16);
  op[0] = o0; op[1] = o1;
}

// ---------------------------------------------------------------------------
// Kernel 2: cast uvqk fp32 -> bf16, pre-swizzled into WMMA B-fragment layout.
// Tile (ko,no): 32 K-rows x 16 N-cols; lane l<16 holds K=ko*32+0..15 at N=no*16+l,
// lane l>=16 holds K=ko*32+16..31.  Tile stored as 32 lanes x 16 contiguous bf16.
// ---------------------------------------------------------------------------
__global__ __launch_bounds__(256) void pack_w_kernel(
    const float* __restrict__ w, __bf16* __restrict__ wp) {
  int idx  = blockIdx.x * 256 + threadIdx.x;     // 0 .. 512*2048-1
  int tile = idx >> 9;
  int lane = (idx >> 4) & 31;
  int j    = idx & 15;
  int ko   = tile >> 7;                          // 16 K-tiles of 32 over E
  int no   = tile & 127;                         // 128 N-tiles of 16 over NPROJ
  int k    = ko * 32 + ((lane >> 4) << 4) + j;
  int n    = no * 16 + (lane & 15);
  wp[idx]  = (__bf16)w[(size_t)k * NPROJ + n];
}

// ---------------------------------------------------------------------------
// Kernel 3: projection GEMM  P = Xn[8192x512] * W[512x2048]  (bf16 WMMA, f32 acc)
// Workgroup tile 128x128 (8 waves, each 64x32 = 4x2 accumulators).
// k-loop is software-pipelined: next iteration's fragments are issued before
// this iteration's WMMAs so global latency overlaps matrix work.
// Output split:  u -> fp32, v -> bf16 transposed [b,h,l,s], q/k -> bf16 row-major.
// ---------------------------------------------------------------------------
__global__ __launch_bounds__(256) void proj_gemm_kernel(
    const __bf16* __restrict__ xn, const __bf16* __restrict__ wp,
    float* __restrict__ U, __bf16* __restrict__ Qb,
    __bf16* __restrict__ Kb, __bf16* __restrict__ Vt) {
  int w     = threadIdx.x >> 5;
  int lane  = threadIdx.x & 31;
  int lhalf = lane >> 4, lmod = lane & 15;
  int mbase = blockIdx.x * 128 + (w >> 2) * 64;
  int nbase = blockIdx.y * 128 + (w & 3) * 32;

  const __bf16* wtile = wp + (((size_t)(nbase >> 4)) << 9) + lane * 16;

  auto loadB = [&](int ko, int ni) -> v16bf {
    return *(const v16bf*)(wtile + ((size_t)(ko * 128 + ni)) * 512);
  };
  auto loadA = [&](int ko, int mi) -> v16bf {
    const __bf16* arow = xn + (size_t)(mbase + mi * 16 + lmod) * EMB + ko * 32 + lhalf * 16;
    return cat8(*(const v8bf*)arow, *(const v8bf*)(arow + 8));
  };

  const v8f z = {0.f,0.f,0.f,0.f,0.f,0.f,0.f,0.f};
  v8f acc[4][2];
  #pragma unroll
  for (int mi = 0; mi < 4; ++mi)
    #pragma unroll
    for (int ni = 0; ni < 2; ++ni) acc[mi][ni] = z;

  // ---- software-pipelined k loop (double-buffered fragments) ----
  v16bf aC[4], bC[2], aN[4], bN[2];
  #pragma unroll
  for (int ni = 0; ni < 2; ++ni) bC[ni] = loadB(0, ni);
  #pragma unroll
  for (int mi = 0; mi < 4; ++mi) aC[mi] = loadA(0, mi);

  #pragma unroll
  for (int ko = 0; ko < 16; ++ko) {
    if (ko < 15) {
      #pragma unroll
      for (int ni = 0; ni < 2; ++ni) bN[ni] = loadB(ko + 1, ni);
      #pragma unroll
      for (int mi = 0; mi < 4; ++mi) aN[mi] = loadA(ko + 1, mi);
      if (ko < 14)   // pull the tile after next into L2 (global_prefetch_b8)
        __builtin_prefetch(wtile + ((size_t)((ko + 2) * 128)) * 512, 0, 1);
    }
    #pragma unroll
    for (int mi = 0; mi < 4; ++mi)
      #pragma unroll
      for (int ni = 0; ni < 2; ++ni)
        acc[mi][ni] = wmma_bf16(aC[mi], bC[ni], acc[mi][ni]);
    #pragma unroll
    for (int ni = 0; ni < 2; ++ni) bC[ni] = bN[ni];
    #pragma unroll
    for (int mi = 0; mi < 4; ++mi) aC[mi] = aN[mi];
  }

  // scatter results: region decided per 16-col subtile (uniform branch)
  #pragma unroll
  for (int mi = 0; mi < 4; ++mi) {
    #pragma unroll
    for (int ni = 0; ni < 2; ++ni) {
      int nb = nbase + ni * 16;
      #pragma unroll
      for (int r = 0; r < 8; ++r) {
        float val = acc[mi][ni][r];
        int m = mbase + mi * 16 + r + lhalf * 8;
        int n = nb + lmod;
        if (nb < 512) {                       // u  (fp32, row-major)
          U[(size_t)m * 512 + n] = val;
        } else if (nb < 1024) {               // v  -> transposed [b,h,l,s]
          int f = n - 512, h = f >> 6, ld = f & 63;
          int b = m >> 11, s = m & (SEQ - 1);
          Vt[(((size_t)(b * NH + h) * LDIM + ld)) * SEQ + s] = (__bf16)val;
        } else if (nb < 1536) {               // q  (bf16, row-major [m][h*64+a])
          Qb[(size_t)m * 512 + (n - 1024)] = (__bf16)val;
        } else {                              // k
          Kb[(size_t)m * 512 + (n - 1536)] = (__bf16)val;
        }
      }
    }
  }
}

// ---------------------------------------------------------------------------
// Kernel 4: streaming HSTU attention.  One workgroup per (b, h, 64-row q block).
// 4 waves x 16 q-rows.  Per key block of 64:
//   issue all K-fragments -> 8 QK WMMAs -> issue all V-fragments (latency
//   hidden behind silu/mask VALU + LDS round-trip + barrier) -> 8 PV WMMAs.
// S x S never touches HBM.
// ---------------------------------------------------------------------------
__global__ __launch_bounds__(128) void attn_kernel(
    const __bf16* __restrict__ Qb, const __bf16* __restrict__ Kb,
    const __bf16* __restrict__ Vt, float* __restrict__ AttnOut) {
  __shared__ __bf16 smem[4 * 16 * 64];
  int w     = threadIdx.x >> 5;
  int lane  = threadIdx.x & 31;
  int lhalf = lane >> 4, lmod = lane & 15;
  int qb = blockIdx.x;            // 0..31 query blocks of 64
  int h  = blockIdx.y;            // head
  int b  = blockIdx.z;            // batch
  int qrow0 = qb * 64 + w * 16;

  // Q as A-fragments (two k-steps over d=0..63), held in registers for all kb
  v16bf aQ[2];
  {
    const __bf16* qr = Qb + (size_t)(b * SEQ + qrow0 + lmod) * EMB + h * 64;
    #pragma unroll
    for (int kk = 0; kk < 2; ++kk) {
      v8bf q0 = *(const v8bf*)(qr + kk * 32 + lhalf * 16);
      v8bf q1 = *(const v8bf*)(qr + kk * 32 + 8 + lhalf * 16);
      aQ[kk] = cat8(q0, q1);
    }
  }

  const __bf16* kbase_p = Kb + (size_t)(b * SEQ + lmod) * EMB + h * 64 + lhalf * 16;
  const __bf16* vbase_p = Vt + ((size_t)((b * NH + h) * LDIM + lmod)) * SEQ + lhalf * 16;

  const v8f z = {0.f,0.f,0.f,0.f,0.f,0.f,0.f,0.f};
  v8f oacc[4];
  #pragma unroll
  for (int t = 0; t < 4; ++t) oacc[t] = z;

  __bf16* myS = smem + w * (16 * 64);

  for (int kb = 0; kb <= qb; ++kb) {
    int krow0 = kb * 64;

    // 1) issue all K B-fragments (lane n = key row, contiguous d per lane)
    v16bf bK[4][2];
    #pragma unroll
    for (int t = 0; t < 4; ++t)
      #pragma unroll
      for (int kk = 0; kk < 2; ++kk)
        bK[t][kk] = *(const v16bf*)(kbase_p + (size_t)(krow0 + t * 16) * EMB + kk * 32);

    // 2) S = Q * K^T
    v8f sacc[4];
    #pragma unroll
    for (int t = 0; t < 4; ++t) sacc[t] = z;
    #pragma unroll
    for (int t = 0; t < 4; ++t)
      #pragma unroll
      for (int kk = 0; kk < 2; ++kk)
        sacc[t] = wmma_bf16(aQ[kk], bK[t][kk], sacc[t]);

    // 3) issue all V B-fragments now; latency hides behind silu + LDS bounce
    v16bf bV[2][4];
    #pragma unroll
    for (int kk = 0; kk < 2; ++kk)
      #pragma unroll
      for (int t = 0; t < 4; ++t)
        bV[kk][t] = *(const v16bf*)(vbase_p + (size_t)(t * 16) * SEQ + krow0 + kk * 32);

    // 4) silu + causal mask (only diagonal block), write bf16 S tile to LDS
    bool diag = (kb == qb);
    #pragma unroll
    for (int t = 0; t < 4; ++t) {
      #pragma unroll
      for (int r = 0; r < 8; ++r) {
        float xv = sacc[t][r];
        float sv = xv / (1.0f + __expf(-xv));
        if (diag) {
          int row = qrow0 + r + lhalf * 8;
          int col = krow0 + t * 16 + lmod;
          if (col > row) sv = 0.0f;
        }
        myS[(r + lhalf * 8) * 64 + t * 16 + lmod] = (__bf16)sv;
      }
    }
    __syncthreads();

    // 5) O += P * V : A = S-tile re-read from LDS in A-fragment layout
    #pragma unroll
    for (int kk = 0; kk < 2; ++kk) {
      v8bf p0 = *(const v8bf*)(myS + lmod * 64 + kk * 32 + lhalf * 16);
      v8bf p1 = *(const v8bf*)(myS + lmod * 64 + kk * 32 + 8 + lhalf * 16);
      v16bf aP = cat8(p0, p1);
      #pragma unroll
      for (int t = 0; t < 4; ++t)
        oacc[t] = wmma_bf16(aP, bV[kk][t], oacc[t]);
    }
    __syncthreads();
  }

  // store attention output tile [16 x 64] (fp32) into [b,s,h*64+l]
  #pragma unroll
  for (int t = 0; t < 4; ++t) {
    #pragma unroll
    for (int r = 0; r < 8; ++r) {
      int m = qrow0 + r + lhalf * 8;
      AttnOut[(size_t)(b * SEQ + m) * EMB + h * 64 + t * 16 + lmod] = oacc[t][r];
    }
  }
}

// ---------------------------------------------------------------------------
// Kernel 5: output LayerNorm * u + residual.  One wave per row.
// ---------------------------------------------------------------------------
__global__ __launch_bounds__(256) void out_kernel(
    const float* __restrict__ x, const float* __restrict__ at,
    const float* __restrict__ U, const float* __restrict__ g,
    const float* __restrict__ be, float* __restrict__ out) {
  int row  = (blockIdx.x * 256 + threadIdx.x) >> 5;
  int lane = threadIdx.x & 31;
  const float* ap = at + (size_t)row * EMB + lane * 16;
  float v[16];
  float s1 = 0.f, s2 = 0.f;
  #pragma unroll
  for (int i = 0; i < 4; ++i) {
    v4f t = *(const v4f*)(ap + i * 4);
    #pragma unroll
    for (int j = 0; j < 4; ++j) { float f = t[j]; v[i*4+j] = f; s1 += f; s2 += f*f; }
  }
  #pragma unroll
  for (int off = 16; off > 0; off >>= 1) {
    s1 += __shfl_xor(s1, off, 32);
    s2 += __shfl_xor(s2, off, 32);
  }
  float mu   = s1 * (1.f / EMB);
  float rstd = rsqrtf(s2 * (1.f / EMB) - mu * mu + 1e-6f);
  const float* gp = g  + lane * 16;
  const float* bp = be + lane * 16;
  const float* up = U  + (size_t)row * EMB + lane * 16;
  const float* xp = x  + (size_t)row * EMB + lane * 16;
  float* op = out + (size_t)row * EMB + lane * 16;
  #pragma unroll
  for (int i = 0; i < 4; ++i) {
    v4f gg = *(const v4f*)(gp + i * 4);
    v4f bb = *(const v4f*)(bp + i * 4);
    v4f uu = *(const v4f*)(up + i * 4);
    v4f xx = *(const v4f*)(xp + i * 4);
    v4f oo;
    #pragma unroll
    for (int j = 0; j < 4; ++j)
      oo[j] = xx[j] + ((v[i*4+j] - mu) * rstd * gg[j] + bb[j]) * uu[j];
    *(v4f*)(op + i * 4) = oo;
  }
}

// ---------------------------------------------------------------------------
extern "C" void kernel_launch(void* const* d_in, const int* in_sizes, int n_in,
                              void* d_out, int out_size, void* d_ws, size_t ws_size,
                              hipStream_t stream) {
  const float* x         = (const float*)d_in[0];
  // d_in[1] = attn_mask (causal, known analytically -> unused)
  const float* uvqk      = (const float*)d_in[2];
  const float* in_gamma  = (const float*)d_in[3];
  const float* in_beta   = (const float*)d_in[4];
  const float* out_gamma = (const float*)d_in[5];
  const float* out_beta  = (const float*)d_in[6];
  float* out = (float*)d_out;

  char* ws = (char*)d_ws;
  __bf16* Xn = (__bf16*)(ws);                          // 8 MB  normed x, bf16
  __bf16* Wp = (__bf16*)(ws + ((size_t)8  << 20));     // 2 MB  packed weights
  float*  U  = (float*) (ws + ((size_t)10 << 20));     // 16 MB u gate, fp32
  __bf16* Qb = (__bf16*)(ws + ((size_t)26 << 20));     // 8 MB  q, bf16
  __bf16* Kb = (__bf16*)(ws + ((size_t)34 << 20));     // 8 MB  k, bf16
  __bf16* Vt = (__bf16*)(ws + ((size_t)42 << 20));     // 8 MB  v transposed
  float*  At = (float*) (ws + ((size_t)50 << 20));     // 16 MB attn out, fp32

  ln_cast_kernel<<<BS / 8, 256, 0, stream>>>(x, in_gamma, in_beta, Xn);
  pack_w_kernel<<<(EMB * NPROJ) / 256, 256, 0, stream>>>(uvqk, Wp);
  proj_gemm_kernel<<<dim3(BS / 128, NPROJ / 128), 256, 0, stream>>>(Xn, Wp, U, Qb, Kb, Vt);
  attn_kernel<<<dim3(SEQ / 64, NH, NBAT), 128, 0, stream>>>(Qb, Kb, Vt, At);
  out_kernel<<<BS / 8, 256, 0, stream>>>(x, At, U, out_gamma, out_beta, out);
}
